// CausalSelfAttention_23364622090693
// MI455X (gfx1250) — compile-verified, matrix operand order checked by reading, not run on hardware
//
#include <hip/hip_runtime.h>
#include <hip/hip_bf16.h>

typedef _Float16 v16h __attribute__((ext_vector_type(16)));
typedef _Float16 v8h  __attribute__((ext_vector_type(8)));
typedef float    v8f  __attribute__((ext_vector_type(8)));

#define D_MODEL 1024
#define N_HEADS 16
#define HD      64
#define TSEQ    2048
#define BATCH   4
#define ROWS    (BATCH * TSEQ)   /* 8192 */
#define QKV_N   (3 * D_MODEL)    /* 3072 */

__device__ __forceinline__ v8f wmma16x16x32(v16h a, v16h b, v8f c) {
  // (neg_a, A, neg_b, B, c_mod, C, reuse_a, reuse_b)
  return __builtin_amdgcn_wmma_f32_16x16x32_f16(false, a, false, b, (short)0, c, false, false);
}

// Low 32 bits of a generic LDS pointer = LDS byte offset (CDNA5 ISA 10.2:
// generic LDS addresses carry the aperture in [63:32] and the offset in [31:0]).
__device__ __forceinline__ uint32_t lds_off_of(const void* p) {
  return (uint32_t)(uintptr_t)p;
}

// GLOBAL_LOAD_ASYNC_TO_LDS_B128: per-lane 16B global -> LDS, tracked by ASYNCcnt.
__device__ __forceinline__ void async_copy_b128(uint32_t ldsOff, const void* g) {
  asm volatile("global_load_async_to_lds_b128 %0, %1, off"
               :: "v"(ldsOff), "v"(g) : "memory");
}
__device__ __forceinline__ void wait_async() {
  asm volatile("s_wait_asynccnt 0x0" ::: "memory");
}

// Load a 16x32 (MxK) f16 WMMA A-fragment from an LDS tile with leading
// dimension `ld` halfs, K offset kBase. Per CDNA5 ISA 7.12.2 (16-bit A 16x32):
// lanes 0-15 = rows M=0-15 with K(kh..kh+7) in VGPR0-3 and K(16+kh..) in
// VGPR4-7, where kh = 8*(lane>=16). The same loader serves B fragments when
// the tile is stored transposed [N][K].
__device__ __forceinline__ v16h lds_frag(const _Float16* base, int ld, int kBase, int lane) {
  int r  = lane & 15;
  int kh = (lane >> 4) << 3;   // 0 or 8
  const _Float16* p = base + r * ld + kBase + kh;
  v8h lo = *(const v8h*)(p);
  v8h hi = *(const v8h*)(p + 16);
  v16h f;
#pragma unroll
  for (int i = 0; i < 8; ++i) { f[i] = lo[i]; f[8 + i] = hi[i]; }
  return f;
}

// ---------------- GEMM: C[M,N] = A[M,K] * Bt[N,K]^T  (f16 in, f32 acc) -----
// 128x128 block tile, 256 threads = 8 wave32; each wave owns a 32x64 strip.
// Double-buffered K loop (BK=32) staged with async global->LDS copies.
__global__ __launch_bounds__(256)
void gemm_h16(const _Float16* __restrict__ A, const _Float16* __restrict__ Bt,
              _Float16* __restrict__ Ch, float* __restrict__ Cf,
              const float* __restrict__ bias, int M, int N, int K) {
  __shared__ _Float16 sA[2][128 * 32];
  __shared__ _Float16 sB[2][128 * 32];
  const int tid  = threadIdx.x;
  const int lane = tid & 31;
  const int wave = tid >> 5;
  const int m0 = blockIdx.y * 128;
  const int n0 = blockIdx.x * 128;

  const int wm = (wave & 3) * 32;    // wave row offset within tile
  const int wn = (wave >> 2) * 64;   // wave col offset within tile

  v8f acc[2][4] = {};

  const int trow = tid >> 1;           // 0..127
  const int tcol = (tid & 1) * 16;     // 0 or 16
  const _Float16* gA = A  + (size_t)(m0 + trow) * K + tcol;
  const _Float16* gB = Bt + (size_t)(n0 + trow) * K + tcol;
  const uint32_t la0 = lds_off_of(&sA[0][trow * 32 + tcol]);
  const uint32_t lb0 = lds_off_of(&sB[0][trow * 32 + tcol]);
  const uint32_t la1 = lds_off_of(&sA[1][trow * 32 + tcol]);
  const uint32_t lb1 = lds_off_of(&sB[1][trow * 32 + tcol]);

  // preload k0 = 0 into buffer 0
  async_copy_b128(la0,      gA);
  async_copy_b128(la0 + 16, gA + 8);
  async_copy_b128(lb0,      gB);
  async_copy_b128(lb0 + 16, gB + 8);
  wait_async();
  __syncthreads();

  for (int k0 = 0; k0 < K; k0 += 32) {
    const int cur = (k0 >> 5) & 1;
    if (k0 + 32 < K) {  // prefetch next K-slab into the other buffer
      const _Float16* ga = gA + k0 + 32;
      const _Float16* gb = gB + k0 + 32;
      const uint32_t la = cur ? la0 : la1;
      const uint32_t lb = cur ? lb0 : lb1;
      async_copy_b128(la,      ga);
      async_copy_b128(la + 16, ga + 8);
      async_copy_b128(lb,      gb);
      async_copy_b128(lb + 16, gb + 8);
    }
    v16h a0 = lds_frag(&sA[cur][(wm +  0) * 32], 32, 0, lane);
    v16h a1 = lds_frag(&sA[cur][(wm + 16) * 32], 32, 0, lane);
#pragma unroll
    for (int n = 0; n < 4; ++n) {
      v16h bf = lds_frag(&sB[cur][(wn + n * 16) * 32], 32, 0, lane);
      acc[0][n] = wmma16x16x32(a0, bf, acc[0][n]);
      acc[1][n] = wmma16x16x32(a1, bf, acc[1][n]);
    }
    wait_async();     // this wave's prefetch landed in LDS
    __syncthreads();  // everyone done reading cur / writing nxt
  }

  // C/D layout: element i of v8f -> row = i + 8*(lane>>4), col = lane&15
  const int colLane = lane & 15;
  const int rowOff  = (lane >> 4) * 8;
#pragma unroll
  for (int mi = 0; mi < 2; ++mi)
#pragma unroll
    for (int n = 0; n < 4; ++n)
#pragma unroll
      for (int i = 0; i < 8; ++i) {
        int row = m0 + wm + mi * 16 + rowOff + i;
        int col = n0 + wn + n * 16 + colLane;
        if (Ch) Ch[(size_t)row * N + col] = (_Float16)acc[mi][n][i];
        else    Cf[(size_t)row * N + col] = acc[mi][n][i] + bias[col];
      }
}

// ---------------- Flash attention (causal), one block per (b, head, 64-q) --
__global__ __launch_bounds__(128)
void flash_attn(const _Float16* __restrict__ qkv, _Float16* __restrict__ Oh) {
  __shared__ _Float16 sQ[64 * 64];
  __shared__ _Float16 sK[64 * 64];       // [key][hd]  == B^T layout for QK^T
  __shared__ _Float16 sVt[64 * 64];      // [hd][key]  == B^T layout for P*V
  __shared__ _Float16 sP[4 * 16 * 64];   // per-wave 16x64 P strip

  const int tid  = threadIdx.x;
  const int lane = tid & 31;
  const int wave = tid >> 5;
  const int qb   = blockIdx.x;
  const int head = blockIdx.y;
  const int b    = blockIdx.z;
  const int q0   = qb * 64;
  const size_t rowBase = (size_t)b * TSEQ;

  { // async-load Q block [64][64]
    int r = tid >> 1, c = (tid & 1) * 32;
    const _Float16* g = qkv + (rowBase + q0 + r) * QKV_N + head * HD + c;
#pragma unroll
    for (int i = 0; i < 32; i += 8)
      async_copy_b128(lds_off_of(&sQ[r * 64 + c + i]), g + i);
  }

  v8f accO[4] = {};
  float mrow[8], lsum[8];
#pragma unroll
  for (int i = 0; i < 8; ++i) { mrow[i] = -1e30f; lsum[i] = 0.f; }

  const int colLane = lane & 15;
  const int rowOff  = (lane >> 4) * 8;
  _Float16* Pw = &sP[wave * 16 * 64];

  for (int j = 0; j <= qb; ++j) {
    __syncthreads();  // prior iteration finished reading sK/sVt
    { // async-load K [key][hd]; manual transposed store of V -> [hd][key]
      int r = tid >> 1, c = (tid & 1) * 32;
      const _Float16* gk = qkv + (rowBase + j * 64 + r) * QKV_N + D_MODEL     + head * HD + c;
      const _Float16* gv = qkv + (rowBase + j * 64 + r) * QKV_N + 2 * D_MODEL + head * HD + c;
#pragma unroll
      for (int i = 0; i < 32; i += 8)
        async_copy_b128(lds_off_of(&sK[r * 64 + c + i]), gk + i);
#pragma unroll
      for (int i = 0; i < 32; ++i)
        sVt[(c + i) * 64 + r] = gv[i];
    }
    wait_async();     // covers K this iter (and Q on j==0)
    __syncthreads();

    // S = Q K^T : this wave's 16 q-rows x 64 keys, Hd=64 => 2 wmma per tile
    v16h qa0 = lds_frag(&sQ[wave * 16 * 64], 64, 0, lane);
    v16h qa1 = lds_frag(&sQ[wave * 16 * 64], 64, 32, lane);
    v8f s[4];
#pragma unroll
    for (int n = 0; n < 4; ++n) {
      v16h kb0 = lds_frag(&sK[n * 16 * 64], 64, 0, lane);
      v16h kb1 = lds_frag(&sK[n * 16 * 64], 64, 32, lane);
      v8f a = {};
      a = wmma16x16x32(qa0, kb0, a);
      a = wmma16x16x32(qa1, kb1, a);
      s[n] = a;
    }

    const bool diag = (j == qb);
#pragma unroll
    for (int n = 0; n < 4; ++n) {
#pragma unroll
      for (int i = 0; i < 8; ++i) {
        float v = s[n][i] * 0.125f;   // 1/sqrt(64)
        if (diag) {
          int key = j * 64 + n * 16 + colLane;
          int qq  = q0 + wave * 16 + rowOff + i;
          if (key > qq) v = -1e30f;
        }
        s[n][i] = v;
      }
    }

    // online softmax; a row's 16 columns live in one 16-lane group
#pragma unroll
    for (int i = 0; i < 8; ++i) {
      float mx = fmaxf(fmaxf(s[0][i], s[1][i]), fmaxf(s[2][i], s[3][i]));
#pragma unroll
      for (int off = 8; off >= 1; off >>= 1) mx = fmaxf(mx, __shfl_xor(mx, off, 32));
      float mnew  = fmaxf(mrow[i], mx);
      float alpha = __expf(mrow[i] - mnew);
      float rs = 0.f;
#pragma unroll
      for (int n = 0; n < 4; ++n) {
        float p = __expf(s[n][i] - mnew);
        s[n][i] = p;
        rs += p;
      }
#pragma unroll
      for (int off = 8; off >= 1; off >>= 1) rs += __shfl_xor(rs, off, 32);
      lsum[i] = lsum[i] * alpha + rs;
      mrow[i] = mnew;
#pragma unroll
      for (int t = 0; t < 4; ++t) accO[t][i] *= alpha;
    }

    // re-layout P (accumulator layout -> A-fragment layout) via LDS
#pragma unroll
    for (int n = 0; n < 4; ++n)
#pragma unroll
      for (int i = 0; i < 8; ++i)
        Pw[(rowOff + i) * 64 + n * 16 + colLane] = (_Float16)s[n][i];
    __syncthreads();

    // O += P V  (K-dim = 64 keys => 2 wmma per hd-tile)
    v16h pa0 = lds_frag(Pw, 64, 0, lane);
    v16h pa1 = lds_frag(Pw, 64, 32, lane);
#pragma unroll
    for (int t = 0; t < 4; ++t) {
      v16h vb0 = lds_frag(&sVt[t * 16 * 64], 64, 0, lane);
      v16h vb1 = lds_frag(&sVt[t * 16 * 64], 64, 32, lane);
      accO[t] = wmma16x16x32(pa0, vb0, accO[t]);
      accO[t] = wmma16x16x32(pa1, vb1, accO[t]);
    }
  }

  // normalize and store; [B,T,h,Hd] flat == [B,T,D] (matches reference transpose)
#pragma unroll
  for (int t = 0; t < 4; ++t) {
#pragma unroll
    for (int i = 0; i < 8; ++i) {
      float o = accO[t][i] / lsum[i];
      int row = q0 + wave * 16 + rowOff + i;
      int col = head * HD + t * 16 + colLane;
      Oh[(rowBase + row) * D_MODEL + col] = (_Float16)o;
    }
  }
}

// ---------------- precision-conversion helpers ------------------------------
__global__ void f32_to_f16(const float* __restrict__ src, _Float16* __restrict__ dst, int n) {
  int i = blockIdx.x * blockDim.x + threadIdx.x;
  if (i < n) dst[i] = (_Float16)src[i];
}
// src [K][N] f32 -> dst [N][K] f16
__global__ void transpose_f32_to_f16(const float* __restrict__ src, _Float16* __restrict__ dst,
                                     int K, int N) {
  int i = blockIdx.x * blockDim.x + threadIdx.x;
  if (i < K * N) {
    int n = i / K, k = i % K;
    dst[i] = (_Float16)src[(size_t)k * N + n];
  }
}

extern "C" void kernel_launch(void* const* d_in, const int* in_sizes, int n_in,
                              void* d_out, int out_size, void* d_ws, size_t ws_size,
                              hipStream_t stream) {
  (void)in_sizes; (void)n_in; (void)out_size; (void)ws_size;
  const float* x     = (const float*)d_in[0];
  const float* w_qkv = (const float*)d_in[1];
  const float* w_out = (const float*)d_in[2];
  const float* b_out = (const float*)d_in[3];
  float* out = (float*)d_out;

  char* ws = (char*)d_ws;                                       // layout (MB):
  _Float16* Xh    = (_Float16*)(ws);                            // [ 0,16) x f16
  _Float16* WqkvT = (_Float16*)(ws + ((size_t)16 << 20));       // [16,22) w_qkv^T
  _Float16* WoutT = (_Float16*)(ws + ((size_t)22 << 20));       // [22,24) w_out^T
  _Float16* QKVh  = (_Float16*)(ws + ((size_t)24 << 20));       // [24,72) qkv f16
  _Float16* Oh    = (_Float16*)(ws + ((size_t)72 << 20));       // [72,88) attn out f16

  const int nx = ROWS * D_MODEL;
  f32_to_f16<<<(nx + 255) / 256, 256, 0, stream>>>(x, Xh, nx);
  const int nwq = D_MODEL * QKV_N;
  transpose_f32_to_f16<<<(nwq + 255) / 256, 256, 0, stream>>>(w_qkv, WqkvT, D_MODEL, QKV_N);
  const int nwo = D_MODEL * D_MODEL;
  transpose_f32_to_f16<<<(nwo + 255) / 256, 256, 0, stream>>>(w_out, WoutT, D_MODEL, D_MODEL);

  gemm_h16<<<dim3(QKV_N / 128, ROWS / 128), 256, 0, stream>>>(
      Xh, WqkvT, QKVh, nullptr, nullptr, ROWS, QKV_N, D_MODEL);

  flash_attn<<<dim3(TSEQ / 64, N_HEADS, BATCH), 128, 0, stream>>>(QKVh, Oh);

  gemm_h16<<<dim3(D_MODEL / 128, ROWS / 128), 256, 0, stream>>>(
      Oh, WoutT, nullptr, out, b_out, ROWS, D_MODEL, D_MODEL);
}
